// BSplineKANLayer_81106162418065
// MI455X (gfx1250) — compile-verified, matrix operand order checked by reading, not run on hardware
//
#include <hip/hip_runtime.h>
#include <hip/hip_bf16.h>

typedef _Float16 v16h __attribute__((ext_vector_type(16)));
typedef _Float16 v8h  __attribute__((ext_vector_type(8)));
typedef _Float16 v4h  __attribute__((ext_vector_type(4)));
typedef float    v8f  __attribute__((ext_vector_type(8)));
typedef float    v4f  __attribute__((ext_vector_type(4)));

#define IN_F   1024
#define OUT_F  1024
#define NCOEF  8                   // GRID_SIZE + SPLINE_ORDER
#define KSEG   9                   // 8 spline segments + 1 base (silu) segment
#define KTOT   (IN_F * KSEG)       // 9216
#define BATCH  8192

#define BM 128
#define BN 256
#define BK 64                      // input features per K stage
#define LDA 72                     // LDS A-tile row stride in halfs (8-half pad)
#define NTHREADS 512               // 16 waves: 2 (M) x 8 (N)

// ---------------------------------------------------------------------------
// Kernel 1: fuse spline_weight*scaler and base_weight into one f16 matrix
//   Wp[n][j*IN_F + i] = spline_weight[n][i][j] * spline_scaler[n][i]  (j<8)
//   Wp[n][8*IN_F + i] = base_weight[n][i]
// ---------------------------------------------------------------------------
__global__ __launch_bounds__(256)
void kan_prep_weights(const float* __restrict__ bw,
                      const float* __restrict__ sw,
                      const float* __restrict__ sc,
                      _Float16* __restrict__ Wp) {
  const int idx = blockIdx.x * 256 + threadIdx.x;   // flat (n, i)
  const int n = idx >> 10;
  const int i = idx & 1023;
  const float s = sc[idx];
  const float* swp = sw + (size_t)idx * NCOEF;
  _Float16* o = Wp + (size_t)n * KTOT + i;
#pragma unroll
  for (int j = 0; j < NCOEF; ++j)
    o[j * IN_F] = (_Float16)(swp[j] * s);
  o[NCOEF * IN_F] = (_Float16)bw[idx];
}

// ---------------------------------------------------------------------------
// Kernel 2: fused KAN GEMM with ping-pong LDS A-tiles.
// out[b][n] = sum_k A(b,k) * Wp[n][k], K = 9216 in 144 segments of 64.
// A(b, j*IN_F+i) = B3((x[b][i]-grid[j])/h) for j<8, silu(x[b][i]) for j==8.
// Segment s+1 is produced (VALU, co-executes with XDL) while segment s is
// consumed by WMMAs -> one barrier per segment instead of two.
// ---------------------------------------------------------------------------
__global__ __launch_bounds__(NTHREADS)
void kan_gemm(const float* __restrict__ x,
              const _Float16* __restrict__ Wp,
              const float* __restrict__ grid,
              float* __restrict__ out) {
  __shared__ __align__(16) _Float16 abuf[2][BM * LDA];   // 2 x 18 KB

  const int tid  = threadIdx.x;
  const int lane = tid & 31;
  const int w    = tid >> 5;       // wave 0..15
  const int wm   = w & 1;          // wave row (2)
  const int wn   = w >> 1;         // wave col (8)

  const int blockM = blockIdx.y * BM;
  const int blockN = blockIdx.x * BN;

  // uniform-grid parameters (grid[m] = lo + (m-3)h)
  const float g3    = grid[3];
  const float h     = grid[4] - g3;
  const float inv_h = 1.0f / h;

  // cooperative x staging: 512 threads -> 4 cols x 4 rows each of 128x64 tile
  const int xc  = (tid & 15) * 4;  // column base within [0,64)
  const int xr0 = tid >> 4;        // rows xr0 + 32*rr, rr = 0..3

  v8f acc[4][2];
#pragma unroll
  for (int mt = 0; mt < 4; ++mt)
#pragma unroll
    for (int nt = 0; nt < 2; ++nt)
      acc[mt][nt] = (v8f)(0.0f);

  float s[4][4];   // normalized positions s = (x - lo)/h for this thread

  // ---- helpers as lambdas (kept register-resident / inlined) --------------
  auto load_x = [&](int ic) {
#pragma unroll
    for (int rr = 0; rr < 4; ++rr) {
      const int row = xr0 + 32 * rr;
      v4f xv = *(const v4f*)(x + (size_t)(blockM + row) * IN_F + ic + xc);
#pragma unroll
      for (int c = 0; c < 4; ++c)
        s[rr][c] = (xv[c] - g3) * inv_h;
    }
  };

  auto produce = [&](int j, _Float16* buf) {
    const float fj = (float)(3 - j);
#pragma unroll
    for (int rr = 0; rr < 4; ++rr) {
      const int row = xr0 + 32 * rr;
      v4h hv;
#pragma unroll
      for (int c = 0; c < 4; ++c) {
        float v;
        if (j < NCOEF) {
          // B3(y) = (y+^3 - 4(y-1)+^3 + 6(y-2)+^3 - 4(y-3)+^3)/6, y in [0,4]
          float y = fminf(fmaxf(s[rr][c] + fj, 0.0f), 4.0f);
          float p0 = y;
          float p1 = fmaxf(y - 1.0f, 0.0f);
          float p2 = fmaxf(y - 2.0f, 0.0f);
          float p3 = fmaxf(y - 3.0f, 0.0f);
          p0 = p0 * p0 * p0; p1 = p1 * p1 * p1;
          p2 = p2 * p2 * p2; p3 = p3 * p3 * p3;
          v = (p0 - 4.0f * p1 + 6.0f * p2 - 4.0f * p3) * (1.0f / 6.0f);
        } else {
          const float xv = s[rr][c] * h + g3;            // recover x
          v = xv / (1.0f + __expf(-xv));                 // silu
        }
        hv[c] = (_Float16)v;
      }
      *(v4h*)(buf + row * LDA + xc) = hv;                // ds_store_b64
    }
  };

  // ---- prologue: stage x chunk 0, produce segment (ic=0, j=0) -------------
  load_x(0);
  produce(0, abuf[0]);
  __syncthreads();

  int seg = 0;
  for (int ic = 0; ic < IN_F; ic += BK) {
    for (int j = 0; j < KSEG; ++j, ++seg) {
      const int cur = seg & 1;

      // ---- produce next segment into the other buffer (overlaps WMMA) ----
      int nj = j + 1;
      bool hasNext = true;
      if (nj == KSEG) {
        nj = 0;
        const int nic = ic + BK;
        hasNext = (nic < IN_F);
        if (hasNext) {
          load_x(nic);
          __builtin_prefetch(x + (size_t)(blockM + xr0) * IN_F + nic + xc, 0, 0);
        }
      }
      if (hasNext) produce(nj, abuf[1 - cur]);

      // ---- consume current segment: 16 WMMAs per wave ---------------------
      const size_t kOff = (size_t)j * IN_F + ic;
      const int arow0 = wm * 64 + (lane & 15);
      const int ko    = (lane >> 4) * 8;                 // half-wave K offset
      const _Float16* cbuf = abuf[cur];
#pragma unroll
      for (int ks = 0; ks < 2; ++ks) {
        const int kb = ks * 32;
        v16h afrag[4];
#pragma unroll
        for (int mt = 0; mt < 4; ++mt) {
          const _Float16* p = cbuf + (arow0 + mt * 16) * LDA + kb + ko;
          v8h lo = *(const v8h*)p;                       // ds_load_b128
          v8h hi = *(const v8h*)(p + 16);
#pragma unroll
          for (int t = 0; t < 8; ++t) { afrag[mt][t] = lo[t]; afrag[mt][t + 8] = hi[t]; }
        }
#pragma unroll
        for (int nt = 0; nt < 2; ++nt) {
          const int col = blockN + wn * 32 + nt * 16 + (lane & 15);
          const _Float16* q = Wp + (size_t)col * KTOT + kOff + kb + ko;
          v8h lo = *(const v8h*)q;                       // global_load_b128 (L2)
          v8h hi = *(const v8h*)(q + 16);
          v16h bfrag;
#pragma unroll
          for (int t = 0; t < 8; ++t) { bfrag[t] = lo[t]; bfrag[t + 8] = hi[t]; }
#pragma unroll
          for (int mt = 0; mt < 4; ++mt)
            acc[mt][nt] = __builtin_amdgcn_wmma_f32_16x16x32_f16(
                false, afrag[mt], false, bfrag, (short)0, acc[mt][nt], false, false);
        }
      }

      __syncthreads();   // single barrier per segment (ping-pong handoff)
    }
  }

  // ---- epilogue: scatter C fragments (VGPR r -> M = r + 8*(lane>=16)) ------
  const int orow0 = blockM + wm * 64 + (lane >> 4) * 8;
  const int ocol  = blockN + wn * 32 + (lane & 15);
#pragma unroll
  for (int mt = 0; mt < 4; ++mt)
#pragma unroll
    for (int nt = 0; nt < 2; ++nt)
#pragma unroll
      for (int r = 0; r < 8; ++r)
        out[(size_t)(orow0 + mt * 16 + r) * OUT_F + ocol + nt * 16] = acc[mt][nt][r];
}

// ---------------------------------------------------------------------------
extern "C" void kernel_launch(void* const* d_in, const int* in_sizes, int n_in,
                              void* d_out, int out_size, void* d_ws, size_t ws_size,
                              hipStream_t stream) {
  (void)in_sizes; (void)n_in; (void)out_size; (void)ws_size;
  const float* x    = (const float*)d_in[0];
  const float* bw   = (const float*)d_in[1];
  const float* sw   = (const float*)d_in[2];
  const float* sc   = (const float*)d_in[3];
  const float* grid = (const float*)d_in[4];
  float* out = (float*)d_out;
  _Float16* Wp = (_Float16*)d_ws;   // 1024*9216*2 = 18.9 MB fused weights

  kan_prep_weights<<<(OUT_F * IN_F) / 256, 256, 0, stream>>>(bw, sw, sc, Wp);

  dim3 gemm_grid(OUT_F / BN, BATCH / BM);   // 4 x 64 workgroups, 16 waves each
  kan_gemm<<<gemm_grid, NTHREADS, 0, stream>>>(x, Wp, grid, out);
}
